// ReformerLM_68427418959943
// MI455X (gfx1250) — matrix operand order repaired and checked by reading, not compiled, not run on hardware
//
#include <hip/hip_runtime.h>
#include <hip/hip_bf16.h>
#include <math.h>

typedef _Float16 f16;
typedef __attribute__((ext_vector_type(16))) _Float16 v16h;
typedef __attribute__((ext_vector_type(8)))  float    v8f;

#define DEPTH   4
#define HEADS   8
#define DH      64
#define NHASH   4
#define DIM     512
#define BATCH   2
#define SEQ     2048
#define ROWS    (BATCH*SEQ)        /* 4096 */
#define BH      (BATCH*HEADS)      /* 16   */
#define NB      32                 /* SEQ/BUCKET */
#define NCHUNK  128                /* NHASH*NB */
#define SORTN   8192               /* NHASH*SEQ */
#define FFD     2048

// -------------------- WMMA helpers --------------------
__device__ __forceinline__ v8f wmma_f16(v16h a, v16h b, v8f c) {
  return __builtin_amdgcn_wmma_f32_16x16x32_f16(false, a, false, b, (short)0, c, false, false);
}
// A fragment (16x32 f16): lane L: m=L%16, khalf=L/16; halves q = K[khalf*16+q].
// `row` points at the 32-wide K slice of row m.
__device__ __forceinline__ v16h load_a_frag(const f16* row, int khalf) {
  v16h a;
  const f16* p0 = row + khalf * 16;
#pragma unroll
  for (int q = 0; q < 16; q++) a[q] = p0[q];
  return a;
}
// B fragment (32x16 f16) from n-major (transposed) LDS: lane L: n=L%16, khalf=L/16;
// halves q = B[khalf*16+q][n] -> contiguous in n-major storage.
__device__ __forceinline__ v16h load_b_frag(const f16* nrow, int khalf) {
  v16h b;
  const f16* p = nrow + khalf * 16;
#pragma unroll
  for (int q = 0; q < 16; q++) b[q] = p[q];
  return b;
}

// -------------------- elementwise / small kernels --------------------
__global__ void zero_i_kernel(int* p, int n) {
  int i = blockIdx.x * 256 + threadIdx.x;
  if (i < n) p[i] = 0;
}

__global__ void f32_to_f16_kernel(const float* __restrict__ in, f16* __restrict__ out, int n) {
  int i = blockIdx.x * 256 + threadIdx.x;
  if (i < n) out[i] = (f16)in[i];
}

__global__ void embed_kernel(const int* __restrict__ x, const float* __restrict__ emb,
                             float* __restrict__ x1, float* __restrict__ x2) {
  size_t i = (size_t)blockIdx.x * 256 + threadIdx.x;   // ROWS*DIM
  int row = (int)(i >> 9);
  int d   = (int)(i & 511);
  int tok = x[row];
  float v = emb[(size_t)tok * DIM + d];
  x1[i] = v; x2[i] = v;
}

// LayerNorm: one wave per 512-wide row; writes f16 (feeds WMMA GEMM A-side)
__global__ __launch_bounds__(256) void ln_kernel(const float* __restrict__ X,
                                                 const float* __restrict__ s,
                                                 const float* __restrict__ b,
                                                 f16* __restrict__ Y) {
  int row  = blockIdx.x * 8 + (threadIdx.x >> 5);
  int lane = threadIdx.x & 31;
  const float* x = X + (size_t)row * DIM;
  float sum = 0.f, sq = 0.f;
  for (int d = lane; d < DIM; d += 32) { float v = x[d]; sum += v; sq += v * v; }
  for (int off = 16; off; off >>= 1) {
    sum += __shfl_xor(sum, off, 32);
    sq  += __shfl_xor(sq,  off, 32);
  }
  float mu  = sum * (1.0f / DIM);
  float var = sq * (1.0f / DIM) - mu * mu;
  float inv = rsqrtf(var + 1e-5f);
  f16* y = Y + (size_t)row * DIM;
  for (int d = lane; d < DIM; d += 32) y[d] = (f16)((x[d] - mu) * inv * s[d] + b[d]);
}

// LN of (x1+x2)*0.5 (final norm before pooling) -> f32
__global__ __launch_bounds__(256) void ln_avg_kernel(const float* __restrict__ X1,
                                                     const float* __restrict__ X2,
                                                     const float* __restrict__ s,
                                                     const float* __restrict__ b,
                                                     float* __restrict__ Y) {
  int row  = blockIdx.x * 8 + (threadIdx.x >> 5);
  int lane = threadIdx.x & 31;
  const float* p1 = X1 + (size_t)row * DIM;
  const float* p2 = X2 + (size_t)row * DIM;
  float sum = 0.f, sq = 0.f;
  for (int d = lane; d < DIM; d += 32) {
    float v = (p1[d] + p2[d]) * 0.5f; sum += v; sq += v * v;
  }
  for (int off = 16; off; off >>= 1) {
    sum += __shfl_xor(sum, off, 32);
    sq  += __shfl_xor(sq,  off, 32);
  }
  float mu  = sum * (1.0f / DIM);
  float var = sq * (1.0f / DIM) - mu * mu;
  float inv = rsqrtf(var + 1e-5f);
  float* y = Y + (size_t)row * DIM;
  for (int d = lane; d < DIM; d += 32) {
    float v = (p1[d] + p2[d]) * 0.5f;
    y[d] = (v - mu) * inv * s[d] + b[d];
  }
}

// deterministic mean over t: pooled[b][d] = mean_t ln[b*SEQ+t][d]
__global__ void pool_reduce_kernel(const float* __restrict__ lnx, float* __restrict__ pooled) {
  int i = blockIdx.x * 256 + threadIdx.x;   // BATCH*DIM = 1024
  if (i >= BATCH * DIM) return;
  int bi = i >> 9, d = i & 511;
  float acc = 0.f;
  const float* p = lnx + (size_t)bi * SEQ * DIM + d;
  for (int t = 0; t < SEQ; t++) acc += p[(size_t)t * DIM];
  pooled[i] = acc * (1.0f / SEQ);
}

__global__ void head1_kernel(const float* __restrict__ pooled, const float* __restrict__ Wf1,
                             const float* __restrict__ bf1, float* __restrict__ hbuf) {
  int bi = blockIdx.x;       // 2
  int j  = threadIdx.x;      // 256
  float acc = bf1[j];
  const float* p = pooled + bi * DIM;
  for (int d = 0; d < DIM; d++) acc += p[d] * Wf1[d * 256 + j];
  hbuf[bi * 256 + j] = fmaxf(acc, 0.0f);
}

__global__ void head2_kernel(const float* __restrict__ hbuf, const float* __restrict__ Wout,
                             const float* __restrict__ bout, float* __restrict__ out) {
  int bi = threadIdx.x;
  if (bi >= BATCH) return;
  float acc = bout[0];
  for (int j = 0; j < 256; j++) acc += hbuf[bi * 256 + j] * Wout[j];
  out[bi] = acc;
}

// ---------- WMMA GEMM (f16 in): C[M,N] = act(A[M,K] @ B[K,N] + bias) (+=/f16 out) ----
// A tile staged via CDNA5 async global->LDS DMA (ASYNCcnt); B tile via manual
// transpose (needs n-major scatter).  Double-buffered LDS, one barrier per K-tile.
#define GF_BIAS   1
#define GF_GELU   2
#define GF_ACC    4
#define GF_F16OUT 8

__global__ __launch_bounds__(256) void gemm_kernel(const f16* __restrict__ A,
                                                   const f16* __restrict__ B,
                                                   const float* __restrict__ bias,
                                                   void* __restrict__ Cv,
                                                   int M, int N, int K, int flags) {
  __shared__ __align__(16) f16 As[2][64][32];     // row-major A tiles (double buffered)
  __shared__ __align__(16) f16 BsT[2][64][32];    // n-major B tiles  (BsT[n][k])
  int m0 = blockIdx.y * 64;
  int n0 = blockIdx.x * 64;
  int tid  = threadIdx.x;
  int lane = tid & 31, w = tid >> 5;
  int mt = w & 3;                // m subtile 0..3
  int ng = w >> 2;               // n-tile group 0..1 -> ntiles ng*2 + {0,1}
  v8f acc0 = {0,0,0,0,0,0,0,0};
  v8f acc1 = {0,0,0,0,0,0,0,0};
  int mrow  = mt * 16 + (lane & 15);
  int khalf = lane >> 4;
  int nk = K >> 5;

  // A: 2048 halves = 64x32, 16 B per thread, raw copy -> async DMA to LDS
  auto loadTileA = [&](int kt, int buf) {
    int k0 = kt << 5;
    int idx = tid * 8;
    int mm = idx >> 5, kk = idx & 31;
    unsigned lds_off = (unsigned)(size_t)&As[buf][mm][kk];  // flat low 32b == LDS offset
    const f16* gp = A + (size_t)(m0 + mm) * K + k0 + kk;
    asm volatile("global_load_async_to_lds_b128 %0, %1, off"
                 :: "v"(lds_off), "v"(gp) : "memory");
  };
  // B: 2048 halves = 32x64, vector global load + n-major LDS scatter (transpose)
  auto loadTileB = [&](int kt, int buf) {
    int k0 = kt << 5;
    int idx = tid * 8;
    int kk = idx >> 6, nn = idx & 63;
    uint4 vB = *(const uint4*)(B + (size_t)(k0 + kk) * N + n0 + nn);
    const f16* hp = (const f16*)&vB;
#pragma unroll
    for (int q = 0; q < 8; q++) BsT[buf][nn + q][kk] = hp[q];
  };

  loadTileA(0, 0);
  loadTileB(0, 0);
  for (int kt = 0; kt < nk; kt++) {
    int cur = kt & 1;
    asm volatile("s_wait_asynccnt 0" ::: "memory");  // my async writes to `cur` done
    __syncthreads();                                 // whole tile visible to all waves
    if (kt + 1 < nk) { loadTileA(kt + 1, cur ^ 1); loadTileB(kt + 1, cur ^ 1); }
    v16h a  = load_a_frag(&As[cur][mrow][0], khalf);
    v16h b0 = load_b_frag(&BsT[cur][(ng * 2 + 0) * 16 + (lane & 15)][0], khalf);
    v16h b1 = load_b_frag(&BsT[cur][(ng * 2 + 1) * 16 + (lane & 15)][0], khalf);
    acc0 = wmma_f16(a, b0, acc0);
    acc1 = wmma_f16(a, b1, acc1);
  }
  int mb = m0 + mt * 16 + (lane >> 4) * 8;
  int nA = n0 + (ng * 2 + 0) * 16 + (lane & 15);
  int nB = n0 + (ng * 2 + 1) * 16 + (lane & 15);
#pragma unroll
  for (int r = 0; r < 8; r++) {
    int row = mb + r;
    float vA = acc0[r], vB = acc1[r];
    if (flags & GF_BIAS) { vA += bias[nA]; vB += bias[nB]; }
    if (flags & GF_GELU) {
      vA = 0.5f * vA * (1.0f + erff(vA * 0.70710678118654752f));
      vB = 0.5f * vB * (1.0f + erff(vB * 0.70710678118654752f));
    }
    size_t iA = (size_t)row * N + nA;
    size_t iB = (size_t)row * N + nB;
    if (flags & GF_F16OUT) {
      f16* C = (f16*)Cv;
      C[iA] = (f16)vA; C[iB] = (f16)vB;
    } else {
      float* C = (float*)Cv;
      if (flags & GF_ACC) { C[iA] += vA; C[iB] += vB; }
      else                { C[iA]  = vA; C[iB]  = vB; }
    }
  }
}

// -------------------- rotary + head split --------------------
// pqk/pv: [ROWS, DIM] f32  ->  qkr/vh: [BH, SEQ, DH] f32 (rotary applied to qk)
__global__ void rotary_kernel(const float* __restrict__ pqk, const float* __restrict__ pv,
                              float* __restrict__ qkr, float* __restrict__ vh) {
  size_t i = (size_t)blockIdx.x * 256 + threadIdx.x;  // BH*SEQ*DH
  int d = (int)(i & 63);
  size_t bt = i >> 6;
  int t  = (int)(bt & 2047);
  int bh = (int)(bt >> 11);
  int b = bh >> 3, hh = bh & 7;
  size_t base = ((size_t)(b * SEQ + t)) * DIM + hh * DH;
  float xv = pqk[base + d];
  float pr = pqk[base + (d ^ 1)];
  float e   = (float)(d & ~1) * (1.0f / DH);
  float inv = expf(-9.210340371976184f * e);   // 10000^-e
  float ang = (float)t * inv;
  float sn = sinf(ang), cs = cosf(ang);
  float out = (d & 1) ? (xv * cs + pr * sn) : (xv * cs - pr * sn);
  qkr[i] = out;
  vh[i]  = pv[base + d];
}

// -------------------- LSH bucketing --------------------
__global__ void bucket_kernel(const float* __restrict__ qkr, const float* __restrict__ rot,
                              int* __restrict__ buckets) {
  int i = blockIdx.x * 256 + threadIdx.x;   // BH*NHASH*SEQ = 131072
  int t    = i & 2047;
  int hash = (i >> 11) & 3;
  int bh   = i >> 13;
  const float* q = qkr + ((size_t)bh * SEQ + t) * DH;
  float rv[16];
#pragma unroll
  for (int j = 0; j < 16; j++) rv[j] = 0.f;
  for (int f = 0; f < DH; f++) {
    float qv = q[f];
    const float* rp = rot + ((size_t)f * NHASH + hash) * 16;
#pragma unroll
    for (int j = 0; j < 16; j++) rv[j] += qv * rp[j];
  }
  float best = rv[0]; int bi = 0;
#pragma unroll
  for (int j = 1; j < 16; j++) if (rv[j] > best) { best = rv[j]; bi = j; }
#pragma unroll
  for (int j = 0; j < 16; j++) if (-rv[j] > best) { best = -rv[j]; bi = 16 + j; }
  buckets[(size_t)bh * SORTN + hash * SEQ + t] = bi + hash * NB;
}

__global__ void hist_kernel(const int* __restrict__ buckets, int* __restrict__ counts) {
  int i = blockIdx.x * 256 + threadIdx.x;   // BH*SORTN
  int bh = i >> 13;
  atomicAdd(&counts[bh * NCHUNK + buckets[i]], 1);
}

__global__ __launch_bounds__(128) void scan_kernel(const int* __restrict__ counts,
                                                   int* __restrict__ offsets) {
  __shared__ int tmp[NCHUNK];
  int bh = blockIdx.x, tid = threadIdx.x;
  int own = counts[bh * NCHUNK + tid];
  tmp[tid] = own;
  __syncthreads();
  for (int off = 1; off < NCHUNK; off <<= 1) {
    int v = (tid >= off) ? tmp[tid - off] : 0;
    __syncthreads();
    tmp[tid] += v;
    __syncthreads();
  }
  offsets[bh * NCHUNK + tid] = tmp[tid] - own;   // exclusive
}

// stable scatter: one wave per (bh, hash); t processed in order -> stable within bucket
__global__ __launch_bounds__(32) void scatter_kernel(const int* __restrict__ buckets,
                                                     const int* __restrict__ offsets,
                                                     int* __restrict__ st,
                                                     int* __restrict__ undo) {
  int bh   = blockIdx.x >> 2;
  int hash = blockIdx.x & 3;
  int lane = threadIdx.x;
  __shared__ int cnt[NB];
  cnt[lane] = 0;
  __syncthreads();
  for (int ch = 0; ch < SEQ / 32; ch++) {
    int t = ch * 32 + lane;
    int i = hash * SEQ + t;
    int bucket = buckets[(size_t)bh * SORTN + i];
    int local  = bucket & 31;
    int base   = cnt[local];
    __syncthreads();
    int rank = 0, total = 0;
    for (int k = 0; k < 32; k++) {
      int bl = __shfl(local, k, 32);
      if (bl == local) { total++; if (k < lane) rank++; }
    }
    int pos = offsets[bh * NCHUNK + bucket] + base + rank;
    st[(size_t)bh * SORTN + pos] = t;
    undo[(size_t)bh * SORTN + i] = pos;
    if (rank == total - 1) cnt[local] = base + total;
    __syncthreads();
  }
}

__global__ void gather_kernel(const float* __restrict__ qkr, const float* __restrict__ vh,
                              const int* __restrict__ st,
                              float* __restrict__ sqk, float* __restrict__ sv) {
  size_t i = (size_t)blockIdx.x * 256 + threadIdx.x;  // BH*SORTN*DH
  int d = (int)(i & 63);
  size_t bj = i >> 6;
  int j  = (int)(bj & (SORTN - 1));
  int bh = (int)(bj >> 13);
  int t = st[(size_t)bh * SORTN + j];
  size_t src = ((size_t)bh * SEQ + t) * DH + d;
  sqk[i] = qkr[src];
  sv[i]  = vh[src];
}

// -------------------- chunked attention (WMMA) --------------------
__global__ __launch_bounds__(256) void attn_kernel(const float* __restrict__ sqk,
                                                   const float* __restrict__ sv,
                                                   const int* __restrict__ st,
                                                   float* __restrict__ so,
                                                   float* __restrict__ slog) {
  int c  = blockIdx.x;          // chunk 0..127
  int bh = blockIdx.y;          // 0..15
  int pc = (c + NCHUNK - 1) & (NCHUNK - 1);
  __shared__ f16 bqh[64][64];    // queries (A for dots)
  __shared__ f16 bkh[128][64];   // normalized keys, row-major == n-major B for dots
  __shared__ f16 bvT[64][128];   // values transposed (n-major B for bo)
  __shared__ f16 dbuf[64][128];  // dots then probabilities
  __shared__ float rnorm[128];
  __shared__ int tq[64], tk[128];
  int tid = threadIdx.x, lane = tid & 31, w = tid >> 5;
  const float* SQ = sqk + (size_t)bh * SORTN * DH;
  const float* SV = sv  + (size_t)bh * SORTN * DH;
  const int*   ST = st  + (size_t)bh * SORTN;
  if (tid < 128) {
    int src = (tid < 64) ? (c * 64 + tid) : (pc * 64 + (tid - 64));
    float s = 0.f;
    for (int d = 0; d < DH; d++) { float v = SQ[(size_t)src * DH + d]; s += v * v; }
    rnorm[tid] = fmaxf(sqrtf(s), 1e-12f);
    tk[tid] = ST[src];
    if (tid < 64) tq[tid] = ST[c * 64 + tid];
  }
  __syncthreads();
  for (int idx = tid; idx < 128 * 64; idx += 256) {
    int j = idx >> 6, d = idx & 63;
    int src = (j < 64) ? (c * 64 + j) : (pc * 64 + (j - 64));
    float v = SQ[(size_t)src * DH + d];
    bkh[j][d] = (f16)(v / rnorm[j]);
    bvT[d][j] = (f16)SV[(size_t)src * DH + d];
    if (j < 64) bqh[j][d] = (f16)v;
  }
  __syncthreads();

  // dots = bq @ bk^T : M=64 N=128 K=64
  int mt  = w & 3;
  int nb4 = (w >> 2) * 4;
  int mrow  = mt * 16 + (lane & 15);
  int khalf = lane >> 4;
  v8f acc[4];
#pragma unroll
  for (int u = 0; u < 4; u++) { v8f z = {0,0,0,0,0,0,0,0}; acc[u] = z; }
#pragma unroll
  for (int ks = 0; ks < 2; ks++) {
    v16h a = load_a_frag(&bqh[mrow][ks * 32], khalf);
#pragma unroll
    for (int u = 0; u < 4; u++) {
      v16h b = load_b_frag(&bkh[(nb4 + u) * 16 + (lane & 15)][ks * 32], khalf);
      acc[u] = wmma_f16(a, b, acc[u]);
    }
  }
  int mb = mt * 16 + (lane >> 4) * 8;
#pragma unroll
  for (int u = 0; u < 4; u++) {
    int n = (nb4 + u) * 16 + (lane & 15);
    int tkn = tk[n];
#pragma unroll
    for (int r = 0; r < 8; r++) {
      int m = mb + r;
      float v = acc[u][r] * 0.125f;          // dh^-0.5
      if (tq[m] == tkn) v = -5.0e4f;         // self-attn mask
      dbuf[m][n] = (f16)v;
    }
  }
  __syncthreads();
  if (tid < 64) {
    float mx = -3.0e38f;
    for (int j = 0; j < 128; j++) { float v = (float)dbuf[tid][j]; mx = fmaxf(mx, v); }
    float s = 0.f;
    for (int j = 0; j < 128; j++) s += expf((float)dbuf[tid][j] - mx);
    float lse = logf(s) + mx;
    slog[(size_t)bh * SORTN + c * 64 + tid] = lse;
    for (int j = 0; j < 128; j++)
      dbuf[tid][j] = (f16)expf((float)dbuf[tid][j] - lse);
  }
  __syncthreads();

  // bo = p @ bv : M=64 N=64 K=128
  int ng = (w >> 2) * 2;
  v8f o0 = {0,0,0,0,0,0,0,0};
  v8f o1 = {0,0,0,0,0,0,0,0};
#pragma unroll
  for (int ks = 0; ks < 4; ks++) {
    v16h a  = load_a_frag(&dbuf[mrow][ks * 32], khalf);
    v16h b0 = load_b_frag(&bvT[(ng + 0) * 16 + (lane & 15)][ks * 32], khalf);
    v16h b1 = load_b_frag(&bvT[(ng + 1) * 16 + (lane & 15)][ks * 32], khalf);
    o0 = wmma_f16(a, b0, o0);
    o1 = wmma_f16(a, b1, o1);
  }
  float* SO = so + ((size_t)bh * SORTN + c * 64) * DH;
#pragma unroll
  for (int r = 0; r < 8; r++) {
    int m = mb + r;
    SO[(size_t)m * DH + (ng + 0) * 16 + (lane & 15)] = o0[r];
    SO[(size_t)m * DH + (ng + 1) * 16 + (lane & 15)] = o1[r];
  }
}

// -------------------- combine hash rounds (writes f16 for Wo GEMM A-side) -----------
__global__ __launch_bounds__(256) void combine_kernel(const float* __restrict__ so,
                                                      const float* __restrict__ slog,
                                                      const int* __restrict__ undo,
                                                      f16* __restrict__ o_cat) {
  int gid  = blockIdx.x * 8 + (threadIdx.x >> 5);  // over BH*SEQ
  int lane = threadIdx.x & 31;
  int bh = gid >> 11;
  int t  = gid & 2047;
  const int* U = undo + (size_t)bh * SORTN;
  int   ui[NHASH];
  float lg[NHASH];
  float mx = -3.0e38f;
#pragma unroll
  for (int h = 0; h < NHASH; h++) {
    ui[h] = U[h * SEQ + t];
    lg[h] = slog[(size_t)bh * SORTN + ui[h]];
    mx = fmaxf(mx, lg[h]);
  }
  float s = 0.f;
#pragma unroll
  for (int h = 0; h < NHASH; h++) s += expf(lg[h] - mx);
  float lse = logf(s) + mx;
  float wgt[NHASH];
#pragma unroll
  for (int h = 0; h < NHASH; h++) wgt[h] = expf(lg[h] - lse);
  int b = bh >> 3, hh = bh & 7;
  f16* dst = o_cat + ((size_t)(b * SEQ + t)) * DIM + hh * DH;
  for (int d = lane; d < DH; d += 32) {
    float o = 0.f;
#pragma unroll
    for (int h = 0; h < NHASH; h++)
      o += wgt[h] * so[((size_t)bh * SORTN + ui[h]) * DH + d];
    dst[d] = (f16)o;
  }
}

// -------------------- host orchestration --------------------
extern "C" void kernel_launch(void* const* d_in, const int* in_sizes, int n_in,
                              void* d_out, int out_size, void* d_ws, size_t ws_size,
                              hipStream_t stream) {
  const int*   x         = (const int*)  d_in[0];
  const float* token_emb = (const float*)d_in[1];
  const float* ln1_s     = (const float*)d_in[2];
  const float* ln1_b     = (const float*)d_in[3];
  const float* Wqk       = (const float*)d_in[4];
  const float* Wv        = (const float*)d_in[5];
  const float* Wo        = (const float*)d_in[6];
  const float* bo        = (const float*)d_in[7];
  const float* ln2_s     = (const float*)d_in[8];
  const float* ln2_b     = (const float*)d_in[9];
  const float* W1        = (const float*)d_in[10];
  const float* b1        = (const float*)d_in[11];
  const float* W2        = (const float*)d_in[12];
  const float* b2        = (const float*)d_in[13];
  const float* rotations = (const float*)d_in[14];
  const float* lnf_s     = (const float*)d_in[15];
  const float* lnf_b     = (const float*)d_in[16];
  const float* Wf1       = (const float*)d_in[17];
  const float* bf1       = (const float*)d_in[18];
  const float* Wout      = (const float*)d_in[19];
  const float* bout      = (const float*)d_in[20];
  float* out = (float*)d_out;

  // workspace carving
  char* p = (char*)d_ws;
  auto carve = [&](size_t bytes) -> char* {
    char* r = p;
    p += (bytes + 255) & ~(size_t)255;
    return r;
  };
  const size_t RD  = (size_t)ROWS * DIM * sizeof(float);      // 8 MB
  const size_t HSD = (size_t)BH * SEQ * DH * sizeof(float);   // 8 MB
  const size_t SSD = (size_t)BH * SORTN * DH * sizeof(float); // 32 MB
  float* x1    = (float*)carve(RD);
  float* x2    = (float*)carve(RD);
  float* xn    = (float*)carve(RD);                            // final LN (f32)
  float* pq    = (float*)carve(RD);
  float* pv    = (float*)carve(RD);
  float* qkr   = (float*)carve(HSD);
  float* vh    = (float*)carve(HSD);
  float* sqk   = (float*)carve(SSD);
  float* svb   = (float*)carve(SSD);
  float* sob   = (float*)carve(SSD);
  float* slg   = (float*)carve((size_t)BH * SORTN * sizeof(float));
  float* pool  = (float*)carve((size_t)BATCH * DIM * sizeof(float));
  float* hbuf  = (float*)carve((size_t)BATCH * 256 * sizeof(float));
  // f16 activations
  f16* xn_h    = (f16*)carve((size_t)ROWS * DIM * sizeof(f16));
  f16* h1_h    = (f16*)carve((size_t)ROWS * FFD * sizeof(f16));
  f16* ocat_h  = (f16*)carve((size_t)ROWS * DIM * sizeof(f16));
  // f16 weights (converted once per launch)
  const int NQK = DEPTH * DIM * DIM;   // 1,048,576
  const int NW1 = DEPTH * DIM * FFD;   // 4,194,304
  f16* wqk_h   = (f16*)carve((size_t)NQK * sizeof(f16));
  f16* wv_h    = (f16*)carve((size_t)NQK * sizeof(f16));
  f16* wo_h    = (f16*)carve((size_t)NQK * sizeof(f16));
  f16* w1_h    = (f16*)carve((size_t)NW1 * sizeof(f16));
  f16* w2_h    = (f16*)carve((size_t)NW1 * sizeof(f16));
  // int scratch
  int* buckets = (int*)carve((size_t)BH * SORTN * sizeof(int));
  int* stb     = (int*)carve((size_t)BH * SORTN * sizeof(int));
  int* undo    = (int*)carve((size_t)BH * SORTN * sizeof(int));
  int* counts  = (int*)carve((size_t)BH * NCHUNK * sizeof(int));
  int* offs    = (int*)carve((size_t)BH * NCHUNK * sizeof(int));

  // weight conversion (f32 -> f16), once per launch
  f32_to_f16_kernel<<<(NQK + 255) / 256, 256, 0, stream>>>(Wqk, wqk_h, NQK);
  f32_to_f16_kernel<<<(NQK + 255) / 256, 256, 0, stream>>>(Wv,  wv_h,  NQK);
  f32_to_f16_kernel<<<(NQK + 255) / 256, 256, 0, stream>>>(Wo,  wo_h,  NQK);
  f32_to_f16_kernel<<<(NW1 + 255) / 256, 256, 0, stream>>>(W1,  w1_h,  NW1);
  f32_to_f16_kernel<<<(NW1 + 255) / 256, 256, 0, stream>>>(W2,  w2_h,  NW1);

  // embedding -> both reversible streams
  embed_kernel<<<ROWS * DIM / 256, 256, 0, stream>>>(x, token_emb, x1, x2);

  for (int l = 0; l < DEPTH; l++) {
    const f16* wqk = wqk_h + (size_t)l * DIM * DIM;
    const f16* wv  = wv_h  + (size_t)l * DIM * DIM;
    const f16* wo  = wo_h  + (size_t)l * DIM * DIM;
    const float* bol = bo  + (size_t)l * DIM;
    const f16* w1  = w1_h  + (size_t)l * DIM * FFD;
    const float* b1l = b1  + (size_t)l * FFD;
    const f16* w2  = w2_h  + (size_t)l * FFD * DIM;
    const float* b2l = b2  + (size_t)l * DIM;
    const float* rot = rotations + (size_t)l * DH * NHASH * (NB / 2);

    // --- attention sub-block: x1 += Attn(LN(x2)) ---
    ln_kernel<<<ROWS / 8, 256, 0, stream>>>(x2, ln1_s + l * DIM, ln1_b + l * DIM, xn_h);
    gemm_kernel<<<dim3(DIM / 64, ROWS / 64), 256, 0, stream>>>(xn_h, wqk, nullptr, pq,
                                                               ROWS, DIM, DIM, 0);
    gemm_kernel<<<dim3(DIM / 64, ROWS / 64), 256, 0, stream>>>(xn_h, wv, nullptr, pv,
                                                               ROWS, DIM, DIM, 0);
    rotary_kernel<<<BH * SEQ * DH / 256, 256, 0, stream>>>(pq, pv, qkr, vh);

    zero_i_kernel<<<(BH * NCHUNK + 255) / 256, 256, 0, stream>>>(counts, BH * NCHUNK);
    bucket_kernel<<<BH * NHASH * SEQ / 256, 256, 0, stream>>>(qkr, rot, buckets);
    hist_kernel<<<BH * SORTN / 256, 256, 0, stream>>>(buckets, counts);
    scan_kernel<<<BH, NCHUNK, 0, stream>>>(counts, offs);
    scatter_kernel<<<BH * NHASH, 32, 0, stream>>>(buckets, offs, stb, undo);
    gather_kernel<<<BH * SORTN * DH / 256, 256, 0, stream>>>(qkr, vh, stb, sqk, svb);
    attn_kernel<<<dim3(NCHUNK, BH), 256, 0, stream>>>(sqk, svb, stb, sob, slg);
    combine_kernel<<<BH * SEQ / 8, 256, 0, stream>>>(sob, slg, undo, ocat_h);
    gemm_kernel<<<dim3(DIM / 64, ROWS / 64), 256, 0, stream>>>(ocat_h, wo, bol, x1,
                                                               ROWS, DIM, DIM,
                                                               GF_BIAS | GF_ACC);

    // --- FF sub-block: x2 += FF(LN(x1)) ---
    ln_kernel<<<ROWS / 8, 256, 0, stream>>>(x1, ln2_s + l * DIM, ln2_b + l * DIM, xn_h);
    gemm_kernel<<<dim3(FFD / 64, ROWS / 64), 256, 0, stream>>>(xn_h, w1, b1l, h1_h,
                                                               ROWS, FFD, DIM,
                                                               GF_BIAS | GF_GELU | GF_F16OUT);
    gemm_kernel<<<dim3(DIM / 64, ROWS / 64), 256, 0, stream>>>(h1_h, w2, b2l, x2,
                                                               ROWS, DIM, FFD,
                                                               GF_BIAS | GF_ACC);
  }

  // final LN over (x1+x2)/2 (xn f32), deterministic mean-pool, MLP head
  ln_avg_kernel<<<ROWS / 8, 256, 0, stream>>>(x1, x2, lnf_s, lnf_b, xn);
  pool_reduce_kernel<<<(BATCH * DIM + 255) / 256, 256, 0, stream>>>(xn, pool);
  head1_kernel<<<BATCH, 256, 0, stream>>>(pool, Wf1, bf1, hbuf);
  head2_kernel<<<1, 32, 0, stream>>>(hbuf, Wout, bout, out);
}
